// MASKSecondary_62732292325832
// MI455X (gfx1250) — matrix operand order repaired and checked by reading, not compiled, not run on hardware
//
#include <hip/hip_runtime.h>

typedef unsigned short u16;
typedef __bf16 v16bf __attribute__((ext_vector_type(16)));
typedef float v8f __attribute__((ext_vector_type(8)));
typedef unsigned int v8u __attribute__((ext_vector_type(8)));

// ---- problem constants ----
#define B_ 2
#define L_ 512
#define E_ 1024
#define H_ 8
#define FF_ 1024
#define C_ 64
#define NB_ 16
#define D_ 128                      // head dim
#define LL_ ((long long)L_ * L_)    // 262144
#define ME_ ((size_t)B_ * L_ * E_)  // 1M tokens*E

// ---------------- bf16 helpers ----------------
__device__ __forceinline__ u16 f2bf(float f) {
  unsigned u = __builtin_bit_cast(unsigned, f);
  unsigned r = (u + 0x7FFFu + ((u >> 16) & 1u)) >> 16;
  return (u16)r;
}
__device__ __forceinline__ float bf2f(u16 s) {
  return __builtin_bit_cast(float, (unsigned)s << 16);
}

// ---------------- CDNA5 async copy global -> LDS ----------------
// Flat-address low 32 bits are the wave-relative LDS byte address (ISA 10.2).
__device__ __forceinline__ unsigned lds_of(const void* p) {
  return (unsigned)(size_t)p;
}
// GLOBAL_LOAD_ASYNC_TO_LDS_B128: 16 bytes per lane, memory -> LDS, ASYNCcnt.
__device__ __forceinline__ void async_b128(unsigned lds_addr, const void* gsrc) {
  asm volatile("global_load_async_to_lds_b128 %0, %1, off"
               :
               : "v"(lds_addr), "v"(gsrc)
               : "memory");
}
// Async loads complete in order; each wave issues exactly 4 per K-tile, so
// waiting ASYNCcnt<=4 guarantees the previous tile's transfers are in LDS.
__device__ __forceinline__ void wait_async_le4() {
  asm volatile("s_wait_asynccnt 4" ::: "memory");
}
__device__ __forceinline__ void wait_async_0() {
  asm volatile("s_wait_asynccnt 0" ::: "memory");
}

// ---------------- WMMA fragment loads from LDS ----------------
// A-matrix 16x32 bf16 layout: lane l -> m = l&15, kLo = (l<16)?0:8,
// halves 0..7 = K kLo..kLo+7, halves 8..15 = K kLo+16..kLo+23.
__device__ __forceinline__ v16bf frag_a(const u16* row, int lane) {
  int kLo = (lane < 16) ? 0 : 8;
  uint4 p0 = *(const uint4*)(row + kLo);
  uint4 p1 = *(const uint4*)(row + kLo + 16);
  v8u u = {p0.x, p0.y, p0.z, p0.w, p1.x, p1.y, p1.z, p1.w};
  return __builtin_bit_cast(v16bf, u);
}
// B-matrix 32x16 bf16: lane l -> n = l&15, kB = (l<16)?0:8,
// halves 0..7 = K kB..kB+7, halves 8..15 = K kB+16..kB+23. row points at Bs[n][0] (rows are "n", K contiguous).
__device__ __forceinline__ v16bf frag_b(const u16* row, int lane) {
  int kB = (lane < 16) ? 0 : 8;
  uint4 p0 = *(const uint4*)(row + kB);
  uint4 p1 = *(const uint4*)(row + kB + 16);
  v8u u = {p0.x, p0.y, p0.z, p0.w, p1.x, p1.y, p1.z, p1.w};
  return __builtin_bit_cast(v16bf, u);
}

// ---------------- generic NT GEMM: C = A[M,K] * Bt[N,K]^T ----------------
// EPI: 0 = bf16 store (+optional bias)        (QKV proj, ctx)
//      1 = bf16 store, bias+ReLU              (FFN1, fc1)
//      2 = bf16 store transposed -> vT[B,E,L] (v projection)
//      3 = attention logits fp32 (+rel bias, masks)
//      4 = fp32 store, bias + residual        (pre-LayerNorm)
//      5 = fp32 store, bias                   (fc2 -> out)
struct GemmP {
  const u16* A; const u16* Bt;
  const float* bias; float* Cf; u16* Cb;
  const float* resid; const int* bucket; const float* rel; const float* mask;
  int M, N, K, lda, ldb, ldc, heads;
  long long sAb, sAh, sBb, sBh, sCb, sCh;
};

template <int EPI>
__global__ __launch_bounds__(128) void gemm_nt(GemmP p) {
  __shared__ __align__(16) u16 As[2][64][32];
  __shared__ __align__(16) u16 Bs[2][64][32];
  __shared__ __align__(16) u16 Dm[16];  // 32B sink for OOB async lanes
  const int m0 = blockIdx.y * 64, n0 = blockIdx.x * 64;
  const int z = blockIdx.z;
  const int bb = z / p.heads, hh = z - bb * p.heads;
  const u16* A = p.A + bb * p.sAb + hh * p.sAh;
  const u16* Bt = p.Bt + bb * p.sBb + hh * p.sBh;
  const int t = threadIdx.x, lane = t & 31, wv = t >> 5;
  const int mr = t >> 1, half = t & 1;  // each thread moves 16 bf16 per matrix
  const unsigned dummyOff = lds_of(&Dm[0]);
  const bool boob = (n0 + mr >= p.N);

  auto issue = [&](int buf, int k0) {
    const u16* ga = A + (size_t)(m0 + mr) * p.lda + k0 + half * 16;
    unsigned la = lds_of(&As[buf][mr][half * 16]);
    async_b128(la, ga);
    async_b128(la + 16, ga + 8);
    const u16* gb = boob ? p.Bt : (Bt + (size_t)(n0 + mr) * p.ldb + k0 + half * 16);
    unsigned lb = boob ? dummyOff : lds_of(&Bs[buf][mr][half * 16]);
    async_b128(lb, gb);
    async_b128(lb + 16, gb + 8);
    if (boob) {  // zero-fill real slot via DS store (ordered by __syncthreads)
      uint4 zz = make_uint4(0u, 0u, 0u, 0u);
      *(uint4*)&Bs[buf][mr][half * 16] = zz;
      *(uint4*)&Bs[buf][mr][half * 16 + 8] = zz;
    }
  };

  v8f acc[4] = {};
  const int ntile = p.K >> 5;
  issue(0, 0);
  for (int it = 0; it < ntile; ++it) {
    const int cur = it & 1;
    if (it + 1 < ntile) {
      issue(cur ^ 1, (it + 1) << 5);  // overlap next tile with this compute
      wait_async_le4();
    } else {
      wait_async_0();
    }
    __syncthreads();
    v16bf af = frag_a(&As[cur][wv * 16 + (lane & 15)][0], lane);
#pragma unroll
    for (int nt = 0; nt < 4; nt++) {
      v16bf bfg = frag_b(&Bs[cur][nt * 16 + (lane & 15)][0], lane);
      acc[nt] = __builtin_amdgcn_wmma_f32_16x16x32_bf16(
          false, af, false, bfg, (short)0, acc[nt], false, false);
    }
    __syncthreads();
  }

  // epilogue: acc[nt][i] -> (m = m0+wv*16+i+(lane>=16?8:0), n = n0+nt*16+(lane&15))
  const size_t cb = (size_t)bb * p.sCb + (size_t)hh * p.sCh;
#pragma unroll
  for (int nt = 0; nt < 4; nt++) {
    int n = n0 + nt * 16 + (lane & 15);
    if (n >= p.N) continue;
#pragma unroll
    for (int i = 0; i < 8; i++) {
      int m = m0 + wv * 16 + i + ((lane >= 16) ? 8 : 0);
      float v = acc[nt][i];
      if constexpr (EPI == 0) {
        if (p.bias) v += p.bias[n];
        p.Cb[cb + (size_t)m * p.ldc + n] = f2bf(v);
      } else if constexpr (EPI == 1) {
        v += p.bias[n];
        v = v > 0.f ? v : 0.f;
        p.Cb[cb + (size_t)m * p.ldc + n] = f2bf(v);
      } else if constexpr (EPI == 2) {
        v += p.bias[n];
        int b2 = m >> 9, ml = m & (L_ - 1);
        p.Cb[((size_t)b2 * E_ + n) * L_ + ml] = f2bf(v);  // vT[B,E,L]
      } else if constexpr (EPI == 3) {
        int bk = p.bucket[m * L_ + n];
        v = (v + p.rel[bk * H_ + hh]) * 0.08838834764831845f;  // 1/sqrt(128)
        if (bk == NB_ - 1) v += -1e9f;
        v += p.mask[bb * L_ + n] * -1e9f;
        p.Cf[cb + (size_t)m * p.ldc + n] = v;
      } else if constexpr (EPI == 4) {
        v += p.bias[n] + p.resid[(size_t)m * p.ldc + n];
        p.Cf[(size_t)m * p.ldc + n] = v;
      } else {
        v += p.bias[n];
        p.Cf[(size_t)m * p.ldc + n] = v;
      }
    }
  }
}

// ---------------- implicit-GEMM conv (NHWC bf16, Cout=64) ----------------
// EPI: 0 = acc+bias -> bf16 ; 1 = relu(acc+bias)+xin -> bf16
struct ConvP {
  const u16* in; const u16* wt; const float* bias;
  const u16* xin; u16* out;
  int kH, kW, pad, dil, CinP;
};

template <int EPI>
__global__ __launch_bounds__(128) void conv_ig(ConvP p) {
  __shared__ __align__(16) u16 As[2][64][32];
  __shared__ __align__(16) u16 Bs[2][64][32];
  __shared__ __align__(16) u16 Dm[16];
  const int y = blockIdx.y, b = blockIdx.z, x0 = blockIdx.x * 64;
  const int t = threadIdx.x, lane = t & 31, wv = t >> 5;
  const int mr = t >> 1, half = t & 1;
  const int Ktot = p.kH * p.kW * p.CinP;
  const int cpc = p.CinP >> 5;             // K-chunks per kernel cell
  const int nch = p.kH * p.kW * cpc;       // total K-chunks
  const unsigned dummyOff = lds_of(&Dm[0]);

  auto issue = [&](int buf, int chunk) {
    int cell = chunk / cpc;
    int c0 = (chunk - cell * cpc) << 5;
    int ky = cell / p.kW, kx = cell - ky * p.kW;
    int iy = y + p.dil * ky - p.pad;
    int ix = x0 + mr + p.dil * kx - p.pad;
    bool ok = ((unsigned)iy < (unsigned)L_) && ((unsigned)ix < (unsigned)L_);
    const u16* ga = ok ? (p.in + (((size_t)b * L_ + iy) * L_ + ix) * p.CinP +
                          c0 + half * 16)
                       : p.in;
    unsigned la = ok ? lds_of(&As[buf][mr][half * 16]) : dummyOff;
    async_b128(la, ga);
    async_b128(la + 16, ga + 8);
    if (!ok) {  // halo: zero-fill via DS store
      uint4 zz = make_uint4(0u, 0u, 0u, 0u);
      *(uint4*)&As[buf][mr][half * 16] = zz;
      *(uint4*)&As[buf][mr][half * 16 + 8] = zz;
    }
    const u16* gb = p.wt + (size_t)mr * Ktot + cell * p.CinP + c0 + half * 16;
    unsigned lb = lds_of(&Bs[buf][mr][half * 16]);
    async_b128(lb, gb);
    async_b128(lb + 16, gb + 8);
  };

  v8f acc[4] = {};
  issue(0, 0);
  for (int it = 0; it < nch; ++it) {
    const int cur = it & 1;
    if (it + 1 < nch) {
      issue(cur ^ 1, it + 1);
      wait_async_le4();
    } else {
      wait_async_0();
    }
    __syncthreads();
    v16bf af = frag_a(&As[cur][wv * 16 + (lane & 15)][0], lane);
#pragma unroll
    for (int nt = 0; nt < 4; nt++) {
      v16bf bfg = frag_b(&Bs[cur][nt * 16 + (lane & 15)][0], lane);
      acc[nt] = __builtin_amdgcn_wmma_f32_16x16x32_bf16(
          false, af, false, bfg, (short)0, acc[nt], false, false);
    }
    __syncthreads();
  }
#pragma unroll
  for (int nt = 0; nt < 4; nt++) {
    int n = nt * 16 + (lane & 15);
#pragma unroll
    for (int i = 0; i < 8; i++) {
      int m = wv * 16 + i + ((lane >= 16) ? 8 : 0);
      size_t oidx = (((size_t)b * L_ + y) * L_ + (x0 + m)) * C_ + n;
      float v = acc[nt][i] + p.bias[n];
      if constexpr (EPI == 1) {
        v = v > 0.f ? v : 0.f;
        v += bf2f(p.xin[oidx]);
      }
      p.out[oidx] = f2bf(v);
    }
  }
}

// ---------------- small kernels ----------------
__global__ void bucket_k(int* bucket) {
  int idx = blockIdx.x * 256 + threadIdx.x;
  int i = idx / L_, j = idx - (idx / L_) * L_;
  int d = i > j ? i - j : j - i;
  int bk;
  if (d < NB_ / 2) bk = d;
  else {
    int lg = NB_ / 2 + (int)(log2f((float)d * (2.0f / NB_)) * (float)(NB_ / 2));
    bk = lg < NB_ - 1 ? lg : NB_ - 1;
  }
  bucket[idx] = bk;
}

__global__ void copyf_bf_k(const float* in, float* of, u16* ob, size_t n) {
  size_t idx = (size_t)blockIdx.x * 256 + threadIdx.x;
  if (idx >= n) return;
  float v = in[idx];
  if (of) of[idx] = v;
  ob[idx] = f2bf(v);
}

// out[n*K + k] = in[k*N + n]  (fp32 -> bf16, pre-transposed weights)
__global__ void transpose_bf16_k(const float* in, u16* out, int K, int N) {
  size_t idx = (size_t)blockIdx.x * 256 + threadIdx.x;
  if (idx >= (size_t)K * N) return;
  int k = idx % K;
  size_t n = idx / K;
  out[idx] = f2bf(in[(size_t)k * N + n]);
}

// conv weight OIHW -> wt[n][(ky*kW+kx)*CinP + c], zero-pad c>=Cin
__global__ void convw_k(const float* w, u16* wt, int Cin, int CinP, int kH, int kW) {
  int Ktot = kH * kW * CinP;
  size_t idx = (size_t)blockIdx.x * 256 + threadIdx.x;
  if (idx >= (size_t)C_ * Ktot) return;
  int n = idx / Ktot, r = idx % Ktot;
  int cell = r / CinP, c = r - cell * CinP;
  int ky = cell / kW, kx = cell - ky * kW;
  float v = (c < Cin) ? w[(((size_t)n * Cin + c) * kH + ky) * kW + kx] : 0.f;
  wt[idx] = f2bf(v);
}

__global__ void softmax_k(const float* alog, u16* attw) {
  const int q = blockIdx.x, h = blockIdx.y, b = blockIdx.z;
  const float* row = alog + ((size_t)b * (H_ * 3) + h) * LL_ + (size_t)q * L_;
  u16* orow = attw + (((size_t)b * H_ + h) * L_ + q) * L_;
  __shared__ float red[256];
  int t = threadIdx.x;
  float x0 = row[t], x1 = row[t + 256];
  red[t] = fmaxf(x0, x1);
  __syncthreads();
  for (int s = 128; s; s >>= 1) { if (t < s) red[t] = fmaxf(red[t], red[t + s]); __syncthreads(); }
  float mx = red[0];
  __syncthreads();
  float e0 = __expf(x0 - mx), e1 = __expf(x1 - mx);
  red[t] = e0 + e1;
  __syncthreads();
  for (int s = 128; s; s >>= 1) { if (t < s) red[t] += red[t + s]; __syncthreads(); }
  float inv = 1.f / red[0];
  orow[t] = f2bf(e0 * inv);
  orow[t + 256] = f2bf(e1 * inv);
}

__global__ void ln_k(const float* in, const float* g, const float* be,
                     float* outf, u16* outb) {
  const int r = blockIdx.x;
  const float* row = in + (size_t)r * E_;
  __shared__ float red[256];
  int t = threadIdx.x;
  float v0 = row[t], v1 = row[t + 256], v2 = row[t + 512], v3 = row[t + 768];
  red[t] = v0 + v1 + v2 + v3;
  __syncthreads();
  for (int s = 128; s; s >>= 1) { if (t < s) red[t] += red[t + s]; __syncthreads(); }
  float mean = red[0] * (1.f / E_);
  __syncthreads();
  float d0 = v0 - mean, d1 = v1 - mean, d2 = v2 - mean, d3 = v3 - mean;
  red[t] = d0 * d0 + d1 * d1 + d2 * d2 + d3 * d3;
  __syncthreads();
  for (int s = 128; s; s >>= 1) { if (t < s) red[t] += red[t + s]; __syncthreads(); }
  float rstd = rsqrtf(red[0] * (1.f / E_) + 1e-6f);
  size_t o = (size_t)r * E_;
  float w0 = g[t] * d0 * rstd + be[t];
  float w1 = g[t + 256] * d1 * rstd + be[t + 256];
  float w2 = g[t + 512] * d2 * rstd + be[t + 512];
  float w3 = g[t + 768] * d3 * rstd + be[t + 768];
  outf[o + t] = w0; outf[o + t + 256] = w1; outf[o + t + 512] = w2; outf[o + t + 768] = w3;
  outb[o + t] = f2bf(w0); outb[o + t + 256] = f2bf(w1);
  outb[o + t + 512] = f2bf(w2); outb[o + t + 768] = f2bf(w3);
}

// a[b,c,i,j] = tanh((a_ij + a_ji)/2) in place; also NHWC bf16 copy (Cin pad 32)
__global__ void symtanh_k(float* a, u16* ahwc) {
  size_t idx = (size_t)blockIdx.x * 256 + threadIdx.x;
  int j = idx % L_;
  size_t r = idx / L_;
  int i = r % L_; r /= L_;
  int c = r % 24;
  int b = r / 24;
  if (i > j) return;
  size_t ij = idx;
  size_t ji = ((size_t)(b * 24 + c) * L_ + j) * L_ + i;
  float s = tanhf(0.5f * (a[ij] + a[ji]));
  a[ij] = s; a[ji] = s;
  u16 bv = f2bf(s);
  ahwc[(((size_t)b * L_ + i) * L_ + j) * 32 + c] = bv;
  ahwc[(((size_t)b * L_ + j) * L_ + i) * 32 + c] = bv;
  if (c == 0) {
    for (int cc = 24; cc < 32; cc++) {
      ahwc[(((size_t)b * L_ + i) * L_ + j) * 32 + cc] = 0;
      ahwc[(((size_t)b * L_ + j) * L_ + i) * 32 + cc] = 0;
    }
  }
}

__global__ void zero_k(float* p, int n) {
  int i = blockIdx.x * 128 + threadIdx.x;
  if (i < n) p[i] = 0.f;
}

__global__ void rowsum_k(const u16* in, float* accum) {
  const int y = blockIdx.x, b = blockIdx.y;
  int t = threadIdx.x, c = t & 63, seg = t >> 6;
  float s = 0.f;
  for (int x = seg * 128; x < seg * 128 + 128; x++)
    s += bf2f(in[(((size_t)b * L_ + y) * L_ + x) * C_ + c]);
  __shared__ float red[256];
  red[t] = s;
  __syncthreads();
  if (t < 64) {
    float tot = red[t] + red[t + 64] + red[t + 128] + red[t + 192];
    atomicAdd(&accum[b * C_ + c], tot);
  }
}

__global__ void ca_k(const float* accum, const float* w1, const float* b1,
                     const float* w2, const float* b2, float* scale) {
  const int b = blockIdx.x;
  int t = threadIdx.x;  // 64 threads
  __shared__ float mean[64], y1[8];
  mean[t] = accum[b * 64 + t] * (1.f / ((float)L_ * (float)L_));
  __syncthreads();
  if (t < 8) {
    float s = b1[t];
    for (int c = 0; c < 64; c++) s += w1[t * 64 + c] * mean[c];
    y1[t] = s > 0.f ? s : 0.f;
  }
  __syncthreads();
  float s = b2[t];
  for (int o = 0; o < 8; o++) s += w2[t * 8 + o] * y1[o];
  scale[b * 64 + t] = 1.f / (1.f + __expf(-s));
}

__global__ void pa_k(const u16* res2, const u16* xin, const float* caScale,
                     const float* w1, const float* b1, const float* w2,
                     const float* b2, u16* out) {
  __shared__ float sw1[512], sb1[8], sw2[8], sca[128];
  int t = threadIdx.x;
  for (int i = t; i < 512; i += 256) sw1[i] = w1[i];
  if (t < 8) { sb1[t] = b1[t]; sw2[t] = w2[t]; }
  if (t < 128) sca[t] = caScale[t];
  __syncthreads();
  size_t pix = (size_t)blockIdx.x * 256 + t;
  int b = (int)(pix / (size_t)LL_);
  size_t base = pix * 64;
  float rs[64];
#pragma unroll
  for (int c = 0; c < 64; c++) rs[c] = bf2f(res2[base + c]) * sca[b * 64 + c];
  float z1[8];
#pragma unroll
  for (int o = 0; o < 8; o++) {
    float s = sb1[o];
    for (int c = 0; c < 64; c++) s += sw1[o * 64 + c] * rs[c];
    z1[o] = s > 0.f ? s : 0.f;
  }
  float zz = b2[0];
#pragma unroll
  for (int o = 0; o < 8; o++) zz += sw2[o] * z1[o];
  float zs = 1.f / (1.f + __expf(-zz));
#pragma unroll
  for (int c = 0; c < 64; c++)
    out[base + c] = f2bf(rs[c] * zs + bf2f(xin[base + c]));
}

__global__ void diag_k(const u16* cur, float* diagf, u16* diagb, int cofs) {
  int idx = blockIdx.x * 256 + threadIdx.x;  // B*L*64
  int c = idx & 63, l = (idx >> 6) & (L_ - 1), b = idx >> 15;
  float v = bf2f(cur[(((size_t)b * L_ + l) * L_ + l) * C_ + c]);
  size_t o = ((size_t)b * L_ + l) * 192 + cofs + c;
  diagf[o] = v;
  diagb[o] = f2bf(v);
}

// ---------------- host ----------------
static GemmP mkg(const u16* A, const u16* Bt, const float* bias, float* Cf,
                 u16* Cb, const float* resid, const int* bucket,
                 const float* rel, const float* mask, int M, int N, int K,
                 int lda, int ldb, int ldc, int heads, long long sAb,
                 long long sAh, long long sBb, long long sBh, long long sCb,
                 long long sCh) {
  GemmP p;
  p.A = A; p.Bt = Bt; p.bias = bias; p.Cf = Cf; p.Cb = Cb; p.resid = resid;
  p.bucket = bucket; p.rel = rel; p.mask = mask;
  p.M = M; p.N = N; p.K = K; p.lda = lda; p.ldb = ldb; p.ldc = ldc;
  p.heads = heads; p.sAb = sAb; p.sAh = sAh; p.sBb = sBb; p.sBh = sBh;
  p.sCb = sCb; p.sCh = sCh;
  return p;
}

extern "C" void kernel_launch(void* const* d_in, const int* in_sizes, int n_in,
                              void* d_out, int out_size, void* d_ws,
                              size_t ws_size, hipStream_t stream) {
  (void)in_sizes; (void)n_in; (void)out_size; (void)ws_size;
  const float* x = (const float*)d_in[0];
  const float* mask = (const float*)d_in[2];
  auto EW = [&](int l, int k) { return (const float*)d_in[3 + 17 * l + k]; };
  const float* first_w = (const float*)d_in[54];
  const float* first_b = (const float*)d_in[55];
  auto BW = [&](int p, int i, int k) { return (const float*)d_in[56 + 36 * p + 12 * i + k]; };
  const float* fc1w = (const float*)d_in[164];
  const float* fc1b = (const float*)d_in[165];
  const float* fc2w = (const float*)d_in[166];
  const float* fc2b = (const float*)d_in[167];

  float* dout = (float*)d_out;
  float* diagF = dout;                       // [B,L,192]
  float* outF = dout + 196608;               // [B,L,8]
  float* aReg = dout + 204800;               // [B,24,L,L]

  // workspace carve-out
  char* base = (char*)d_ws;
  size_t off = 0;
  auto alloc = [&](size_t bytes) -> void* {
    off = (off + 255) & ~(size_t)255;
    void* p = base + off;
    off += bytes;
    return p;
  };
  int* bucket = (int*)alloc(LL_ * 4);
  float* hF = (float*)alloc(ME_ * 4);
  u16* hB = (u16*)alloc(ME_ * 2);
  float* t0 = (float*)alloc(ME_ * 4);
  float* aLF = (float*)alloc(ME_ * 4);
  u16* aLB = (u16*)alloc(ME_ * 2);
  u16* qB = (u16*)alloc(ME_ * 2);
  u16* kB = (u16*)alloc(ME_ * 2);
  u16* vT = (u16*)alloc(ME_ * 2);
  u16* attwB = (u16*)alloc((size_t)B_ * H_ * LL_ * 2);
  u16* ctxB = (u16*)alloc(ME_ * 2);
  u16* interB = (u16*)alloc(ME_ * 2);
  u16* wT[3][6];
  for (int l = 0; l < 3; l++)
    for (int j = 0; j < 6; j++) wT[l][j] = (u16*)alloc((size_t)E_ * FF_ * 2);
  u16* fwT = (u16*)alloc((size_t)C_ * 800 * 2);
  u16* bwT[3][3][2];
  int ks[3] = {5, 3, 3}, pads[3] = {2, 2, 1}, dils[3] = {1, 2, 1};
  for (int p = 0; p < 3; p++)
    for (int i = 0; i < 3; i++)
      for (int j = 0; j < 2; j++)
        bwT[p][i][j] = (u16*)alloc((size_t)C_ * (ks[p] * ks[p] * C_) * 2);
  u16* fc1T = (u16*)alloc((size_t)96 * 192 * 2);
  u16* fc2T = (u16*)alloc((size_t)8 * 96 * 2);
  u16* ahwc = (u16*)alloc((size_t)B_ * LL_ * 32 * 2);
  u16* h0 = (u16*)alloc((size_t)B_ * LL_ * C_ * 2);
  u16* cur = (u16*)alloc((size_t)B_ * LL_ * C_ * 2);
  u16* bufA = (u16*)alloc((size_t)B_ * LL_ * C_ * 2);
  u16* bufB = (u16*)alloc((size_t)B_ * LL_ * C_ * 2);
  float* accum = (float*)alloc(B_ * C_ * 4);
  float* caS = (float*)alloc(B_ * C_ * 4);
  u16* diagB = (u16*)alloc((size_t)1024 * 192 * 2);
  u16* fc1o = (u16*)alloc((size_t)1024 * 96 * 2);

  const dim3 blk128(128), blk256(256);

  // ---- setup ----
  bucket_k<<<LL_ / 256, blk256, 0, stream>>>(bucket);
  copyf_bf_k<<<(int)((ME_ + 255) / 256), blk256, 0, stream>>>(x, hF, hB, ME_);
  for (int l = 0; l < 3; l++) {
    transpose_bf16_k<<<4096, blk256, 0, stream>>>(EW(l, 0), wT[l][0], E_, E_);   // wq
    transpose_bf16_k<<<4096, blk256, 0, stream>>>(EW(l, 2), wT[l][1], E_, E_);   // wk
    transpose_bf16_k<<<4096, blk256, 0, stream>>>(EW(l, 4), wT[l][2], E_, E_);   // wv
    transpose_bf16_k<<<4096, blk256, 0, stream>>>(EW(l, 6), wT[l][3], E_, E_);   // wo
    transpose_bf16_k<<<4096, blk256, 0, stream>>>(EW(l, 11), wT[l][4], E_, FF_); // wi
    transpose_bf16_k<<<4096, blk256, 0, stream>>>(EW(l, 13), wT[l][5], FF_, E_); // wo2
  }
  convw_k<<<(64 * 800 + 255) / 256, blk256, 0, stream>>>(first_w, fwT, 24, 32, 5, 5);
  for (int p = 0; p < 3; p++)
    for (int i = 0; i < 3; i++)
      for (int j = 0; j < 2; j++) {
        int kk = ks[p], tot = C_ * kk * kk * C_;
        convw_k<<<(tot + 255) / 256, blk256, 0, stream>>>(BW(p, i, j * 2),
                                                          bwT[p][i][j], C_, C_, kk, kk);
      }
  transpose_bf16_k<<<(96 * 192 + 255) / 256, blk256, 0, stream>>>(fc1w, fc1T, 192, 96);
  transpose_bf16_k<<<(8 * 96 + 255) / 256, blk256, 0, stream>>>(fc2w, fc2T, 96, 8);

  // ---- encoder ----
  for (int l = 0; l < 3; l++) {
    gemm_nt<0><<<dim3(16, 16, 1), blk128, 0, stream>>>(
        mkg(hB, wT[l][0], EW(l, 1), nullptr, qB, nullptr, nullptr, nullptr,
            nullptr, 1024, E_, E_, E_, E_, E_, 1, 0, 0, 0, 0, 0, 0));
    gemm_nt<0><<<dim3(16, 16, 1), blk128, 0, stream>>>(
        mkg(hB, wT[l][1], EW(l, 3), nullptr, kB, nullptr, nullptr, nullptr,
            nullptr, 1024, E_, E_, E_, E_, E_, 1, 0, 0, 0, 0, 0, 0));
    gemm_nt<2><<<dim3(16, 16, 1), blk128, 0, stream>>>(
        mkg(hB, wT[l][2], EW(l, 5), nullptr, vT, nullptr, nullptr, nullptr,
            nullptr, 1024, E_, E_, E_, E_, E_, 1, 0, 0, 0, 0, 0, 0));
    // logits -> a region (batched over b,h)
    gemm_nt<3><<<dim3(8, 8, B_ * H_), blk128, 0, stream>>>(
        mkg(qB, kB, nullptr, aReg + (size_t)l * H_ * LL_, nullptr, nullptr,
            bucket, EW(l, 8), mask, L_, L_, D_, E_, E_, L_, H_,
            (long long)L_ * E_, D_, (long long)L_ * E_, D_, 24ll * LL_, LL_));
    softmax_k<<<dim3(L_, H_, B_), blk256, 0, stream>>>(aReg + (size_t)l * H_ * LL_, attwB);
    // ctx = attw @ v  (NT via vT)
    gemm_nt<0><<<dim3(2, 8, B_ * H_), blk128, 0, stream>>>(
        mkg(attwB, vT, nullptr, nullptr, ctxB, nullptr, nullptr, nullptr,
            nullptr, L_, D_, L_, L_, L_, E_, H_, (long long)H_ * LL_, LL_,
            (long long)E_ * L_, (long long)D_ * L_, (long long)L_ * E_, D_));
    gemm_nt<4><<<dim3(16, 16, 1), blk128, 0, stream>>>(
        mkg(ctxB, wT[l][3], EW(l, 7), t0, nullptr, hF, nullptr, nullptr,
            nullptr, 1024, E_, E_, E_, E_, E_, 1, 0, 0, 0, 0, 0, 0));
    ln_k<<<1024, blk256, 0, stream>>>(t0, EW(l, 9), EW(l, 10), aLF, aLB);
    gemm_nt<1><<<dim3(16, 16, 1), blk128, 0, stream>>>(
        mkg(aLB, wT[l][4], EW(l, 12), nullptr, interB, nullptr, nullptr,
            nullptr, nullptr, 1024, FF_, E_, E_, E_, FF_, 1, 0, 0, 0, 0, 0, 0));
    gemm_nt<4><<<dim3(16, 16, 1), blk128, 0, stream>>>(
        mkg(interB, wT[l][5], EW(l, 14), t0, nullptr, aLF, nullptr, nullptr,
            nullptr, 1024, E_, FF_, FF_, FF_, E_, 1, 0, 0, 0, 0, 0, 0));
    ln_k<<<1024, blk256, 0, stream>>>(t0, EW(l, 15), EW(l, 16), hF, hB);
  }

  // ---- symmetrize + tanh, build NHWC bf16 (Cin padded to 32) ----
  symtanh_k<<<(int)((size_t)B_ * 24 * LL_ / 256), blk256, 0, stream>>>(aReg, ahwc);

  // ---- first conv 5x5 pad2 ----
  {
    ConvP cp; cp.in = ahwc; cp.wt = fwT; cp.bias = first_b; cp.xin = nullptr;
    cp.out = h0; cp.kH = 5; cp.kW = 5; cp.pad = 2; cp.dil = 1; cp.CinP = 32;
    conv_ig<0><<<dim3(L_ / 64, L_, B_), blk128, 0, stream>>>(cp);
  }

  // ---- three residual-block paths (h0 reused, cur overwritten per path) ----
  for (int p = 0; p < 3; p++) {
    const u16* inp = h0;
    for (int i = 0; i < 3; i++) {
      ConvP c1; c1.in = inp; c1.wt = bwT[p][i][0]; c1.bias = BW(p, i, 1);
      c1.xin = inp; c1.out = bufA; c1.kH = ks[p]; c1.kW = ks[p];
      c1.pad = pads[p]; c1.dil = dils[p]; c1.CinP = C_;
      conv_ig<1><<<dim3(L_ / 64, L_, B_), blk128, 0, stream>>>(c1);
      ConvP c2 = c1; c2.in = bufA; c2.wt = bwT[p][i][1]; c2.bias = BW(p, i, 3);
      c2.xin = nullptr; c2.out = bufB;
      conv_ig<0><<<dim3(L_ / 64, L_, B_), blk128, 0, stream>>>(c2);
      zero_k<<<1, blk128, 0, stream>>>(accum, B_ * C_);
      rowsum_k<<<dim3(L_, B_), blk256, 0, stream>>>(bufB, accum);
      ca_k<<<B_, dim3(64), 0, stream>>>(accum, BW(p, i, 4), BW(p, i, 5),
                                        BW(p, i, 6), BW(p, i, 7), caS);
      pa_k<<<(int)((size_t)B_ * LL_ / 256), blk256, 0, stream>>>(
          bufB, inp, caS, BW(p, i, 8), BW(p, i, 9), BW(p, i, 10), BW(p, i, 11), cur);
      inp = cur;
    }
    diag_k<<<256, blk256, 0, stream>>>(cur, diagF, diagB, p * 64);
  }

  // ---- final FCs ----
  gemm_nt<1><<<dim3(2, 16, 1), blk128, 0, stream>>>(
      mkg(diagB, fc1T, fc1b, nullptr, fc1o, nullptr, nullptr, nullptr, nullptr,
          1024, 96, 192, 192, 192, 96, 1, 0, 0, 0, 0, 0, 0));
  gemm_nt<5><<<dim3(1, 16, 1), blk128, 0, stream>>>(
      mkg(fc1o, fc2T, fc2b, outF, nullptr, nullptr, nullptr, nullptr, nullptr,
          1024, 8, 96, 96, 96, 8, 1, 0, 0, 0, 0, 0, 0));
}
